// TemporalContrastiveLoss_15470472200964
// MI455X (gfx1250) — matrix operand order repaired and checked by reading, not compiled, not run on hardware
//
#include <hip/hip_runtime.h>
#include <hip/hip_bf16.h>
#include <math.h>
#include <stdint.h>

typedef __attribute__((ext_vector_type(16))) _Float16 v16h;
typedef __attribute__((ext_vector_type(8)))  _Float16 v8h;
typedef __attribute__((ext_vector_type(8)))  float    v8f;

#define B_       64
#define T_       512
#define D_       1024
#define NM_      5
#define TV_      (T_ - NM_ - 1)      // 506
#define INV_TEMP 10.0f
#define MAXS     10.0f               // scores are dot/TEMP with |dot|<=1 -> bounded by 10

#define SROWS    64                  // s-rows staged per chunk
#define KC       128                 // K extent staged per chunk
#define RS       (KC + 8)            // padded LDS row stride (halves): 68 dwords = 4 mod 64 banks

// ---------------------------------------------------------------------------
// Kernel 1: row L2-normalize fp32 -> fp16 feats (one 256-thread block per row)
// ---------------------------------------------------------------------------
__global__ __launch_bounds__(256) void tcl_normalize(const float* __restrict__ h,
                                                     _Float16* __restrict__ feats) {
  const int row = blockIdx.x;                       // 0 .. B*T-1
  const float* src = h + (size_t)row * D_;
  _Float16* dst = feats + (size_t)row * D_;

  float v[4];
  float ss = 0.f;
#pragma unroll
  for (int i = 0; i < 4; ++i) {
    v[i] = src[threadIdx.x + i * 256];
    ss += v[i] * v[i];
  }
#pragma unroll
  for (int off = 16; off >= 1; off >>= 1) ss += __shfl_xor(ss, off, 32);

  __shared__ float wsum[8];
  if ((threadIdx.x & 31) == 0) wsum[threadIdx.x >> 5] = ss;
  __syncthreads();
  float tot = 0.f;
#pragma unroll
  for (int w = 0; w < 8; ++w) tot += wsum[w];

  const float inv = 1.f / fmaxf(sqrtf(tot), 1e-12f);
#pragma unroll
  for (int i = 0; i < 4; ++i) dst[threadIdx.x + i * 256] = (_Float16)(v[i] * inv);
}

// ---------------------------------------------------------------------------
// Kernel 2: Gram tiles via v_wmma_f32_16x16x32_f16 with CDNA5 async DMA
// staging of the shared B operand into LDS (double-buffered). Each wave owns
// a 32-row t-tile (2 A fragments, 8 accumulators) so every LDS B fragment
// feeds two WMMAs (ds:wmma = 1:1). 8 waves per block share the staged tile.
// grid = (T/32/8, B), block = 256 (8 waves).
// ---------------------------------------------------------------------------
__global__ __launch_bounds__(256) void tcl_scores_lse(const _Float16* __restrict__ feats,
                                                      float* __restrict__ partials) {
  __shared__ _Float16 Bs[2][SROWS * RS];            // 2 x 17 KB, bank-conflict-padded

  const int tid  = threadIdx.x;
  const int lane = tid & 31;
  const int wave = tid >> 5;
  const int b    = blockIdx.y;
  const int tile = blockIdx.x * 8 + wave;           // 0..15 (32-row tiles)
  const int tbase = tile * 32;
  const int half = lane >> 4;                       // 0 or 1
  const int l16  = lane & 15;

  const _Float16* Fb = feats + (size_t)b * T_ * D_;
  // A (16x32 f16): lanes 0-15 hold K {0..7,16..23}, lanes 16-31 K {8..15,24..31}
  // of row tbase+l16 (ISA 7.12.2). Second A tile is 16 rows further down.
  const _Float16* arow0 = Fb + (size_t)(tbase + l16) * D_ + half * 8;
  const _Float16* arow1 = arow0 + (size_t)16 * D_;

  // Staging assignment: thread -> (row, 32-half segment); 4x b128 DMA per thread.
  const int srow = tid & 63;
  const int sseg = tid >> 6;                        // 0..3

  float sumexp[2][8], possum[2][8];
#pragma unroll
  for (int tt = 0; tt < 2; ++tt)
#pragma unroll
    for (int r = 0; r < 8; ++r) { sumexp[tt][r] = 0.f; possum[tt][r] = 0.f; }

  auto stage = [&](int buf, int sbase, int k0) {
    uint32_t lds = (uint32_t)(uintptr_t)&Bs[buf][srow * RS + sseg * 32];
    const _Float16* g = Fb + (size_t)(sbase + srow) * D_ + k0 + sseg * 32;
#pragma unroll
    for (int i = 0; i < 4; ++i) {
      asm volatile("global_load_async_to_lds_b128 %0, %1, off"
                   :: "v"(lds + i * 16), "v"(g + i * 8) : "memory");
    }
  };

  auto aload = [&](const _Float16* ar, int k) -> v16h {
    v8h a0 = *(const v8h*)(ar + k);
    v8h a1 = *(const v8h*)(ar + k + 16);
    v16h a;
#pragma unroll
    for (int i = 0; i < 8; ++i) { a[i] = a0[i]; a[8 + i] = a1[i]; }
    return a;
  };

  for (int sblk = 0; sblk < T_ / SROWS; ++sblk) {
    const int sbase = sblk * SROWS;

    v8f cA0 = {}, cA1 = {}, cA2 = {}, cA3 = {};     // rows tbase..tbase+15
    v8f cB0 = {}, cB1 = {}, cB2 = {}, cB3 = {};     // rows tbase+16..tbase+31

    stage(0, sbase, 0);
    asm volatile("s_wait_asynccnt 0x0" ::: "memory");
    __syncthreads();

    for (int kblk = 0; kblk < D_ / KC; ++kblk) {
      const int cur = kblk & 1;
      if (kblk < D_ / KC - 1) stage(cur ^ 1, sbase, (kblk + 1) * KC);  // prefetch

      const int k0 = kblk * KC;
#pragma unroll
      for (int kk = 0; kk < KC; kk += 32) {
        const v16h a0 = aload(arow0, k0 + kk);
        const v16h a1 = aload(arow1, k0 + kk);

        // B fragments from LDS: lane = column s, lanes 0-15 hold K {0..7,16..23},
        // lanes 16-31 hold K {8..15,24..31} (two 16B ds_load_b128 per lane).
        auto bfrag = [&](int j) -> v16h {
          const _Float16* p = &Bs[cur][(j * 16 + l16) * RS + kk + half * 8];
          v8h b0 = *(const v8h*)p;
          v8h b1 = *(const v8h*)(p + 16);
          v16h bv;
#pragma unroll
          for (int i = 0; i < 8; ++i) { bv[i] = b0[i]; bv[8 + i] = b1[i]; }
          return bv;
        };
        const v16h b0 = bfrag(0), b1 = bfrag(1), b2 = bfrag(2), b3 = bfrag(3);

        cA0 = __builtin_amdgcn_wmma_f32_16x16x32_f16(false, a0, false, b0, (short)0, cA0, false, false);
        cB0 = __builtin_amdgcn_wmma_f32_16x16x32_f16(false, a1, false, b0, (short)0, cB0, false, false);
        cA1 = __builtin_amdgcn_wmma_f32_16x16x32_f16(false, a0, false, b1, (short)0, cA1, false, false);
        cB1 = __builtin_amdgcn_wmma_f32_16x16x32_f16(false, a1, false, b1, (short)0, cB1, false, false);
        cA2 = __builtin_amdgcn_wmma_f32_16x16x32_f16(false, a0, false, b2, (short)0, cA2, false, false);
        cB2 = __builtin_amdgcn_wmma_f32_16x16x32_f16(false, a1, false, b2, (short)0, cB2, false, false);
        cA3 = __builtin_amdgcn_wmma_f32_16x16x32_f16(false, a0, false, b3, (short)0, cA3, false, false);
        cB3 = __builtin_amdgcn_wmma_f32_16x16x32_f16(false, a1, false, b3, (short)0, cB3, false, false);
      }

      asm volatile("s_wait_asynccnt 0x0" ::: "memory");   // prefetch landed
      __syncthreads();                                    // all waves done with 'cur'
    }

    // Per-lane epilogue: fixed-shift exp accumulation, no cross-lane traffic.
    auto epilogue = [&](const v8f& c, int tt, int j) {
      const int s = sbase + j * 16 + l16;
#pragma unroll
      for (int r = 0; r < 8; ++r) {
        const int t = tbase + tt * 16 + half * 8 + r;
        const float sc = c[r] * INV_TEMP;
        const bool live  = (t < TV_);
        const bool ispos = live && (s != t) && (s >= t - 2) && (s <= t + 2);
        const bool isneg = live && (s >= t + NM_);
        possum[tt][r] += ispos ? sc : 0.f;
        sumexp[tt][r] += isneg ? __expf(sc - MAXS) : 0.f;
      }
    };
    epilogue(cA0, 0, 0); epilogue(cA1, 0, 1); epilogue(cA2, 0, 2); epilogue(cA3, 0, 3);
    epilogue(cB0, 1, 0); epilogue(cB1, 1, 1); epilogue(cB2, 1, 2); epilogue(cB3, 1, 3);
  }

  // Single cross-lane reduction per wave, then close the logsumexp.
  float loss = 0.f;
#pragma unroll
  for (int tt = 0; tt < 2; ++tt) {
#pragma unroll
    for (int r = 0; r < 8; ++r) {
      float se = sumexp[tt][r], ps = possum[tt][r];
#pragma unroll
      for (int off = 8; off >= 1; off >>= 1) {
        se += __shfl_xor(se, off, 32);
        ps += __shfl_xor(ps, off, 32);
      }
      const int t = tbase + tt * 16 + half * 8 + r;
      const float cnt = (t == 0) ? 2.f : (t == 1) ? 3.f : 4.f;
      const float pos = ps / cnt;
      if (t < TV_) {
        const float lse = MAXS + __logf(se + __expf(pos - MAXS));
        loss += lse - pos;
      }
    }
  }
  loss += __shfl_xor(loss, 16, 32);                 // combine both half-waves
  if (lane == 0) partials[b * 16 + tile] = loss;
}

// ---------------------------------------------------------------------------
// Kernel 3: deterministic reduction of 1024 partials -> mean loss scalar
// ---------------------------------------------------------------------------
__global__ __launch_bounds__(256) void tcl_reduce(const float* __restrict__ partials,
                                                  float* __restrict__ out) {
  float s = 0.f;
  for (int i = threadIdx.x; i < B_ * 16; i += 256) s += partials[i];
#pragma unroll
  for (int off = 16; off >= 1; off >>= 1) s += __shfl_xor(s, off, 32);
  __shared__ float wsum[8];
  if ((threadIdx.x & 31) == 0) wsum[threadIdx.x >> 5] = s;
  __syncthreads();
  if (threadIdx.x == 0) {
    float tot = 0.f;
    for (int w = 0; w < 8; ++w) tot += wsum[w];
    out[0] = tot / (float)(B_ * TV_);
  }
}

// ---------------------------------------------------------------------------
extern "C" void kernel_launch(void* const* d_in, const int* in_sizes, int n_in,
                              void* d_out, int out_size, void* d_ws, size_t ws_size,
                              hipStream_t stream) {
  (void)in_sizes; (void)n_in; (void)out_size; (void)ws_size;
  const float* hs = (const float*)d_in[0];
  _Float16* feats = (_Float16*)d_ws;                                  // 64 MB fp16
  float* partials = (float*)((char*)d_ws + (size_t)B_ * T_ * D_ * sizeof(_Float16));

  tcl_normalize<<<B_ * T_, 256, 0, stream>>>(hs, feats);
  tcl_scores_lse<<<dim3(T_ / 32 / 8, B_), 256, 0, stream>>>(feats, partials);
  tcl_reduce<<<1, 256, 0, stream>>>(partials, (float*)d_out);
}
